// PointerNet_39676907888277
// MI455X (gfx1250) — matrix operand order repaired and checked by reading, not compiled, not run on hardware
//
#include <hip/hip_runtime.h>

#define BB 256
#define SS 128
#define EE 256
#define HH 512
#define GG 2048  // 4*H

typedef __attribute__((ext_vector_type(16))) _Float16 v16h;
typedef __attribute__((ext_vector_type(8)))  float    v8f;
typedef __attribute__((ext_vector_type(2)))  _Float16 h2;

__device__ __forceinline__ v8f zero8() {
    v8f z = {0.f, 0.f, 0.f, 0.f, 0.f, 0.f, 0.f, 0.f};
    return z;
}

__device__ __forceinline__ float sigmoidf(float x) {
    return 1.0f / (1.0f + __expf(-x));
}

// A-fragment (16x32 f16) from a row-major [16 x ld] tile.
// ISA layout: lanes 0-15 hold M=lane, K={0..7,16..23}; lanes 16-31 hold K={8..15,24..31}.
__device__ __forceinline__ v16h load_a16(const _Float16* base, int ld, int k0, int lane) {
    const int m   = lane & 15;
    const int khi = (lane >> 4) * 8;
    const _Float16* row = base + m * ld + k0 + khi;
    v16h a;
#pragma unroll
    for (int v = 0; v < 8; ++v) {
        const int k = (v < 4) ? (2 * v) : (8 + 2 * v);  // 0,2,4,6,16,18,20,22
        h2 p = *(const h2*)(row + k);
        a[2 * v]     = p.x;
        a[2 * v + 1] = p.y;
    }
    return a;
}

// B-fragment (32x16 f16, K x N) from row-major weight W[N][K] (i.e. B = W^T).
// Lane n holds column N=n0+(lane&15); lanes 0-15 hold K={0..7,16..23}, lanes 16-31 hold K={8..15,24..31}.
__device__ __forceinline__ v16h load_b16(const _Float16* W, int ldk, int n0, int k0, int lane) {
    const int n   = n0 + (lane & 15);
    const int khi = (lane >> 4) * 8;
    const _Float16* row = W + (size_t)n * ldk + k0 + khi;
    v16h b;
#pragma unroll
    for (int v = 0; v < 8; ++v) {
        h2 p = *(const h2*)(row + ((v < 4) ? (2 * v) : (8 + 2 * v)));
        b[2 * v]     = p.x;
        b[2 * v + 1] = p.y;
    }
    return b;
}

__device__ __forceinline__ v8f wmma_f16(v16h a, v16h b, v8f c) {
    return __builtin_amdgcn_wmma_f32_16x16x32_f16(false, a, false, b, (short)0, c, false, false);
}

// ---------------------------------------------------------------------------
// Elementwise helpers
// ---------------------------------------------------------------------------
__global__ void cvt_f16_kernel(const float* __restrict__ in, _Float16* __restrict__ out, int n) {
    int i = blockIdx.x * blockDim.x + threadIdx.x;
    if (i < n) out[i] = (_Float16)in[i];
}

__global__ void add_f32_kernel(const float* __restrict__ a, const float* __restrict__ b,
                               float* __restrict__ out, int n) {
    int i = blockIdx.x * blockDim.x + threadIdx.x;
    if (i < n) out[i] = a[i] + b[i];
}

// emb[b,s,e] = in[b,s,0]*W[e,0] + in[b,s,1]*W[e,1] + bias[e], stored f16
__global__ void embed_kernel(const float* __restrict__ in, const float* __restrict__ W,
                             const float* __restrict__ bias, _Float16* __restrict__ out) {
    int i  = blockIdx.x * blockDim.x + threadIdx.x;  // B*S*E
    int e  = i & (EE - 1);
    int bs = i >> 8;
    float v = in[bs * 2] * W[e * 2] + in[bs * 2 + 1] * W[e * 2 + 1] + bias[e];
    out[i] = (_Float16)v;
}

// ---------------------------------------------------------------------------
// Persistent encoder LSTM: each block owns 16 batch rows for all S steps.
// 512 threads = 16 waves. Two column-group passes per step: in pass cg the
// wave owns hidden columns [16*(w+16*cg), +16) across the 4 gates => acc[4]
// (32 VGPRs) live at a time; c-state lives in VGPRs.
// ---------------------------------------------------------------------------
__global__ __launch_bounds__(512)
void encoder_kernel(const _Float16* __restrict__ embh,   // [B,S,E] f16
                    const _Float16* __restrict__ Wih,    // [4H,E] f16
                    const _Float16* __restrict__ Whh,    // [4H,H] f16
                    const float* __restrict__ bsum,      // [4H] bih+bhh
                    const float* __restrict__ h0,        // [H]
                    const float* __restrict__ c0,        // [H]
                    _Float16* __restrict__ enc_out,      // [B,S,H] f16
                    float* __restrict__ c_state)         // [B,H]
{
    __shared__ _Float16 xh[16][EE + 8];
    __shared__ _Float16 hh[16][HH + 8];

    const int tid  = threadIdx.x;
    const int lane = tid & 31;
    const int w    = tid >> 5;
    const int b0   = blockIdx.x * 16;
    const int nlo  = lane & 15;
    const int mhi  = (lane >> 4) * 8;

    // init h tile (broadcast h0)
    for (int i = tid; i < 16 * HH; i += 512) {
        int m = i >> 9, n = i & (HH - 1);
        hh[m][n] = (_Float16)h0[n];
    }

    float c[2][8];
    float bias[2][4];
#pragma unroll
    for (int cg = 0; cg < 2; ++cg) {
        int n = 16 * (w + 16 * cg) + nlo;
        float cv = c0[n];
#pragma unroll
        for (int r = 0; r < 8; ++r) c[cg][r] = cv;
#pragma unroll
        for (int g = 0; g < 4; ++g) bias[cg][g] = bsum[g * HH + n];
    }
    __syncthreads();

    for (int t = 0; t < SS; ++t) {
        // stage x_t tile into LDS
        for (int i = tid; i < 16 * EE; i += 512) {
            int m = i >> 8, e = i & (EE - 1);
            xh[m][e] = embh[((size_t)(b0 + m) * SS + t) * EE + e];
        }
        __syncthreads();

        float hn[2][8];
#pragma unroll 1
        for (int cg = 0; cg < 2; ++cg) {
            const int ncol = 16 * (w + 16 * cg);
            v8f acc[4];
#pragma unroll
            for (int g = 0; g < 4; ++g) acc[g] = zero8();

            // input contribution: K over E
#pragma unroll 1
            for (int kc = 0; kc < EE; kc += 32) {
                v16h a = load_a16(&xh[0][0], EE + 8, kc, lane);
#pragma unroll
                for (int g = 0; g < 4; ++g) {
                    v16h bf = load_b16(Wih, EE, g * HH + ncol, kc, lane);
                    acc[g] = wmma_f16(a, bf, acc[g]);
                }
            }
            // recurrent contribution: K over H
#pragma unroll 1
            for (int kc = 0; kc < HH; kc += 32) {
                v16h a = load_a16(&hh[0][0], HH + 8, kc, lane);
#pragma unroll
                for (int g = 0; g < 4; ++g) {
                    v16h bf = load_b16(Whh, HH, g * HH + ncol, kc, lane);
                    acc[g] = wmma_f16(a, bf, acc[g]);
                }
            }

            // LSTM cell (torch gate order i,f,g,o); c stays in registers
#pragma unroll
            for (int r = 0; r < 8; ++r) {
                float iv = acc[0][r] + bias[cg][0];
                float fv = acc[1][r] + bias[cg][1];
                float gv = acc[2][r] + bias[cg][2];
                float ov = acc[3][r] + bias[cg][3];
                float cn = sigmoidf(fv) * c[cg][r] + sigmoidf(iv) * tanhf(gv);
                c[cg][r]  = cn;
                hn[cg][r] = sigmoidf(ov) * tanhf(cn);
            }
        }
        __syncthreads();  // all WMMA reads of hh/xh done
#pragma unroll
        for (int cg = 0; cg < 2; ++cg)
#pragma unroll
            for (int r = 0; r < 8; ++r) {
                int m = mhi + r;
                int n = 16 * (w + 16 * cg) + nlo;
                _Float16 hv = (_Float16)hn[cg][r];
                hh[m][n] = hv;
                enc_out[((size_t)(b0 + m) * SS + t) * HH + n] = hv;
            }
        __syncthreads();  // new h visible for next step
    }

    // final cell state for the decoder
#pragma unroll
    for (int cg = 0; cg < 2; ++cg)
#pragma unroll
        for (int r = 0; r < 8; ++r)
            c_state[(size_t)(b0 + mhi + r) * HH + 16 * (w + 16 * cg) + nlo] = c[cg][r];
}

// ---------------------------------------------------------------------------
// enc_proj = enc_out @ W1^T + b1 : [32768,512] x [512,512], one wave per tile
// ---------------------------------------------------------------------------
__global__ __launch_bounds__(256)
void proj_kernel(const _Float16* __restrict__ A,    // [B*S, H] f16
                 const _Float16* __restrict__ W1h,  // [H, H] f16
                 const float* __restrict__ b1,      // [H]
                 float* __restrict__ out)           // [B*S, H] f32
{
    const int lane = threadIdx.x & 31;
    const int w    = threadIdx.x >> 5;
    const int tile = blockIdx.x * 8 + w;
    const int NT   = HH / 16;  // 32
    const int m0   = (tile / NT) * 16;
    const int n0   = (tile % NT) * 16;

    v8f acc = zero8();
#pragma unroll 1
    for (int kc = 0; kc < HH; kc += 32) {
        v16h a = load_a16(A + (size_t)m0 * HH, HH, kc, lane);
        v16h b = load_b16(W1h, HH, n0, kc, lane);
        acc = wmma_f16(a, b, acc);
    }
    const int n   = n0 + (lane & 15);
    const int mhi = (lane >> 4) * 8;
    const float bb = b1[n];
#pragma unroll
    for (int r = 0; r < 8; ++r)
        out[(size_t)(m0 + mhi + r) * HH + n] = acc[r] + bb;
}

// ---------------------------------------------------------------------------
// Persistent decoder: LSTM cell (WMMA) + W2 projection (WMMA) + attention
// scores + masked softmax/argmax (wave-per-row, wave32 shuffles) + gather.
// ---------------------------------------------------------------------------
__global__ __launch_bounds__(512)
void decoder_kernel(const float* __restrict__ inputs,     // [B,S,2]
                    const _Float16* __restrict__ embh,    // [B,S,E] f16
                    const _Float16* __restrict__ enc_out, // [B,S,H] f16
                    const float* __restrict__ enc_proj,   // [B*S,H] f32
                    const float* __restrict__ c_state,    // [B,H]
                    const _Float16* __restrict__ x0h,     // [E] f16
                    const _Float16* __restrict__ Wih,     // dec [4H,E] f16
                    const _Float16* __restrict__ Whh,     // dec [4H,H] f16
                    const float* __restrict__ bsum,       // dec bih+bhh [4H]
                    const _Float16* __restrict__ W2h,     // [H,H] f16
                    const float* __restrict__ b2,         // [H]
                    const float* __restrict__ Vv,         // [H]
                    float* __restrict__ probs_out,        // [S,B,S]
                    int* __restrict__ idx_out)            // [B,S]
{
    __shared__ _Float16 xh[16][EE + 8];
    __shared__ _Float16 hh[16][HH + 8];
    __shared__ _Float16 w2h[16][HH + 8];
    __shared__ float    usc[16][SS];
    __shared__ unsigned char msk[16][SS];
    __shared__ int      sel[16];
    __shared__ float    Vs[HH];

    const int tid  = threadIdx.x;
    const int lane = tid & 31;
    const int w    = tid >> 5;
    const int b0   = blockIdx.x * 16;
    const int nlo  = lane & 15;
    const int mhi  = (lane >> 4) * 8;

    // init: h = encoder final h (= enc_out[:, S-1, :]); x = dec_input0; mask
    for (int i = tid; i < 16 * HH; i += 512) {
        int m = i >> 9, n = i & (HH - 1);
        hh[m][n] = enc_out[((size_t)(b0 + m) * SS + (SS - 1)) * HH + n];
    }
    for (int i = tid; i < 16 * EE; i += 512) {
        int m = i >> 8, e = i & (EE - 1);
        xh[m][e] = x0h[e];
    }
    for (int i = tid; i < 16 * SS; i += 512) {
        int m = i >> 7, s = i & (SS - 1);
        msk[m][s] = inputs[((size_t)(b0 + m) * SS + s) * 2] > 0.0f ? 1 : 0;
    }
    for (int i = tid; i < HH; i += 512) Vs[i] = Vv[i];

    float c[2][8];
    float bias[2][4];
#pragma unroll
    for (int cg = 0; cg < 2; ++cg) {
        int n = 16 * (w + 16 * cg) + nlo;
#pragma unroll
        for (int r = 0; r < 8; ++r)
            c[cg][r] = c_state[(size_t)(b0 + mhi + r) * HH + n];
#pragma unroll
        for (int g = 0; g < 4; ++g) bias[cg][g] = bsum[g * HH + n];
    }
    __syncthreads();

    for (int t = 0; t < SS; ++t) {
        // ---- decoder LSTM cell (two column-group passes) ----
        float hn[2][8];
#pragma unroll 1
        for (int cg = 0; cg < 2; ++cg) {
            const int ncol = 16 * (w + 16 * cg);
            v8f acc[4];
#pragma unroll
            for (int g = 0; g < 4; ++g) acc[g] = zero8();

#pragma unroll 1
            for (int kc = 0; kc < EE; kc += 32) {
                v16h a = load_a16(&xh[0][0], EE + 8, kc, lane);
#pragma unroll
                for (int g = 0; g < 4; ++g) {
                    v16h bf = load_b16(Wih, EE, g * HH + ncol, kc, lane);
                    acc[g] = wmma_f16(a, bf, acc[g]);
                }
            }
#pragma unroll 1
            for (int kc = 0; kc < HH; kc += 32) {
                v16h a = load_a16(&hh[0][0], HH + 8, kc, lane);
#pragma unroll
                for (int g = 0; g < 4; ++g) {
                    v16h bf = load_b16(Whh, HH, g * HH + ncol, kc, lane);
                    acc[g] = wmma_f16(a, bf, acc[g]);
                }
            }
#pragma unroll
            for (int r = 0; r < 8; ++r) {
                float iv = acc[0][r] + bias[cg][0];
                float fv = acc[1][r] + bias[cg][1];
                float gv = acc[2][r] + bias[cg][2];
                float ov = acc[3][r] + bias[cg][3];
                float cn = sigmoidf(fv) * c[cg][r] + sigmoidf(iv) * tanhf(gv);
                c[cg][r]  = cn;
                hn[cg][r] = sigmoidf(ov) * tanhf(cn);
            }
        }
        __syncthreads();
#pragma unroll
        for (int cg = 0; cg < 2; ++cg)
#pragma unroll
            for (int r = 0; r < 8; ++r)
                hh[mhi + r][16 * (w + 16 * cg) + nlo] = (_Float16)hn[cg][r];
        __syncthreads();

        // ---- w2h = h @ W2^T + b2 (two single-tile passes) ----
#pragma unroll 1
        for (int j = 0; j < 2; ++j) {
            const int n0 = 16 * (w + 16 * j);
            v8f a2 = zero8();
#pragma unroll 1
            for (int kc = 0; kc < HH; kc += 32) {
                v16h a  = load_a16(&hh[0][0], HH + 8, kc, lane);
                v16h bf = load_b16(W2h, HH, n0, kc, lane);
                a2 = wmma_f16(a, bf, a2);
            }
            int n = n0 + nlo;
            float bb = b2[n];
#pragma unroll
            for (int r = 0; r < 8; ++r)
                w2h[mhi + r][n] = (_Float16)(a2[r] + bb);
        }
        __syncthreads();

        // ---- attention scores: u[b,s] = sum_n V[n]*tanh(enc_proj + w2h) ----
        for (int p = tid; p < 16 * SS; p += 512) {
            int row = p >> 7, s = p & (SS - 1);
            const float* ep = enc_proj + ((size_t)(b0 + row) * SS + s) * HH;
            float u = 0.0f;
            for (int n = 0; n < HH; ++n)
                u += Vs[n] * tanhf(ep[n] + (float)w2h[row][n]);
            usc[row][s] = msk[row][s] ? u : -10000.0f;
        }
        __syncthreads();

        // ---- masked softmax + argmax: wave w owns batch row w ----
        {
            const int row = w;
            float v4[4];
#pragma unroll
            for (int q = 0; q < 4; ++q) v4[q] = usc[row][lane + 32 * q];
            float mx = fmaxf(fmaxf(v4[0], v4[1]), fmaxf(v4[2], v4[3]));
#pragma unroll
            for (int off = 16; off >= 1; off >>= 1)
                mx = fmaxf(mx, __shfl_xor(mx, off, 32));
            float ev[4], se = 0.0f;
#pragma unroll
            for (int q = 0; q < 4; ++q) {
                ev[q] = __expf(v4[q] - mx);
                se += ev[q];
            }
#pragma unroll
            for (int off = 16; off >= 1; off >>= 1)
                se += __shfl_xor(se, off, 32);
            float inv = 1.0f / se;
#pragma unroll
            for (int q = 0; q < 4; ++q)
                probs_out[(size_t)t * BB * SS + (size_t)(b0 + row) * SS + lane + 32 * q] =
                    ev[q] * inv;
            // argmax (first occurrence of max)
            float bv = v4[0];
            int   bi = lane;
#pragma unroll
            for (int q = 1; q < 4; ++q)
                if (v4[q] > bv) { bv = v4[q]; bi = lane + 32 * q; }
#pragma unroll
            for (int off = 16; off >= 1; off >>= 1) {
                float ov = __shfl_xor(bv, off, 32);
                int   oi = __shfl_xor(bi, off, 32);
                if (ov > bv || (ov == bv && oi < bi)) { bv = ov; bi = oi; }
            }
            if (lane == 0) {
                sel[row]          = bi;
                msk[row][bi]      = 0;
                idx_out[(b0 + row) * SS + t] = bi;
            }
        }
        __syncthreads();

        // ---- next input: x = emb[b, argmax] ----
        for (int i = tid; i < 16 * EE; i += 512) {
            int row = i >> 8, e = i & (EE - 1);
            xh[row][e] = embh[((size_t)(b0 + row) * SS + sel[row]) * EE + e];
        }
        __syncthreads();
    }
}

// ---------------------------------------------------------------------------
extern "C" void kernel_launch(void* const* d_in, const int* in_sizes, int n_in,
                              void* d_out, int out_size, void* d_ws, size_t ws_size,
                              hipStream_t stream) {
    (void)in_sizes; (void)n_in; (void)out_size; (void)ws_size;
    const float* inputs   = (const float*)d_in[0];
    const float* emb_W    = (const float*)d_in[2];
    const float* emb_b    = (const float*)d_in[3];
    const float* enc_Wih  = (const float*)d_in[4];
    const float* enc_Whh  = (const float*)d_in[5];
    const float* enc_bih  = (const float*)d_in[6];
    const float* enc_bhh  = (const float*)d_in[7];
    const float* h0       = (const float*)d_in[8];
    const float* c0       = (const float*)d_in[9];
    const float* dec_in0  = (const float*)d_in[10];
    const float* dec_Wih  = (const float*)d_in[11];
    const float* dec_Whh  = (const float*)d_in[12];
    const float* dec_bih  = (const float*)d_in[13];
    const float* dec_bhh  = (const float*)d_in[14];
    const float* W1       = (const float*)d_in[15];
    const float* b1       = (const float*)d_in[16];
    const float* W2       = (const float*)d_in[17];
    const float* b2       = (const float*)d_in[18];
    const float* V        = (const float*)d_in[19];

    // workspace carve-out (all sizes 256B aligned)
    char* p = (char*)d_ws;
    _Float16* embh    = (_Float16*)p; p += (size_t)BB * SS * EE * 2;   // 16 MB
    _Float16* encout  = (_Float16*)p; p += (size_t)BB * SS * HH * 2;   // 32 MB
    float*    encproj = (float*)p;    p += (size_t)BB * SS * HH * 4;   // 64 MB
    float*    cstate  = (float*)p;    p += (size_t)BB * HH * 4;
    _Float16* eWih    = (_Float16*)p; p += (size_t)GG * EE * 2;
    _Float16* eWhh    = (_Float16*)p; p += (size_t)GG * HH * 2;
    _Float16* dWih    = (_Float16*)p; p += (size_t)GG * EE * 2;
    _Float16* dWhh    = (_Float16*)p; p += (size_t)GG * HH * 2;
    _Float16* W1h     = (_Float16*)p; p += (size_t)HH * HH * 2;
    _Float16* W2h     = (_Float16*)p; p += (size_t)HH * HH * 2;
    float*    bsum_e  = (float*)p;    p += (size_t)GG * 4;
    float*    bsum_d  = (float*)p;    p += (size_t)GG * 4;
    _Float16* x0h     = (_Float16*)p; p += 512;

    // precision staging: f32 -> f16 weights (L2-resident, ~7 MB total)
    cvt_f16_kernel<<<(GG * EE + 255) / 256, 256, 0, stream>>>(enc_Wih, eWih, GG * EE);
    cvt_f16_kernel<<<(GG * HH + 255) / 256, 256, 0, stream>>>(enc_Whh, eWhh, GG * HH);
    cvt_f16_kernel<<<(GG * EE + 255) / 256, 256, 0, stream>>>(dec_Wih, dWih, GG * EE);
    cvt_f16_kernel<<<(GG * HH + 255) / 256, 256, 0, stream>>>(dec_Whh, dWhh, GG * HH);
    cvt_f16_kernel<<<(HH * HH + 255) / 256, 256, 0, stream>>>(W1, W1h, HH * HH);
    cvt_f16_kernel<<<(HH * HH + 255) / 256, 256, 0, stream>>>(W2, W2h, HH * HH);
    cvt_f16_kernel<<<(EE + 255) / 256, 256, 0, stream>>>(dec_in0, x0h, EE);
    add_f32_kernel<<<(GG + 255) / 256, 256, 0, stream>>>(enc_bih, enc_bhh, bsum_e, GG);
    add_f32_kernel<<<(GG + 255) / 256, 256, 0, stream>>>(dec_bih, dec_bhh, bsum_d, GG);

    // embedding
    embed_kernel<<<(BB * SS * EE) / 256, 256, 0, stream>>>(inputs, emb_W, emb_b, embh);

    // encoder LSTM (persistent, 16 batch rows per block)
    encoder_kernel<<<BB / 16, 512, 0, stream>>>(embh, eWih, eWhh, bsum_e, h0, c0,
                                                encout, cstate);

    // enc_proj = enc_out @ W1^T + b1
    proj_kernel<<<(BB * SS / 16) * (HH / 16) / 8, 256, 0, stream>>>(encout, W1h, b1, encproj);

    // decoder LSTM + attention (persistent)
    float* probs = (float*)d_out;
    int*   idxs  = (int*)((char*)d_out + (size_t)SS * BB * SS * sizeof(float));
    decoder_kernel<<<BB / 16, 512, 0, stream>>>(inputs, embh, encout, encproj, cstate, x0h,
                                                dWih, dWhh, bsum_d, W2h, b2, V, probs, idxs);
}
